// TransformerBlock_34909494182091
// MI455X (gfx1250) — matrix operand order repaired and checked by reading, not compiled, not run on hardware
//
#include <hip/hip_runtime.h>

// ---------------------------------------------------------------------------
// BigBird transformer block on MI455X (gfx1250, wave32).
// - All GEMM-shaped math on V_WMMA_F32_16X16X32_BF16 (bf16 in, f32 accum).
// - Attention K/V blocks double-buffered in LDS via the Tensor Data Mover
//   (tensor_load_to_lds, TENSORcnt), overlapped with compute.
// ---------------------------------------------------------------------------

typedef __bf16 bf16;
typedef __bf16 v16bf __attribute__((ext_vector_type(16)));
typedef float  v8f   __attribute__((ext_vector_type(8)));
typedef unsigned int v4u  __attribute__((ext_vector_type(4)));
typedef unsigned int u32x4 __attribute__((ext_vector_type(4)));
typedef int i32x4 __attribute__((ext_vector_type(4)));
typedef int i32x8 __attribute__((ext_vector_type(8)));

#define B_  2
#define M_  4096
#define H_  12
#define D_  64
#define E_  768
#define FF_ 3072
#define NB_ 64
#define R_  3
#define NEG_ (-10000.0f)
#define BM_ (B_*M_)

union FragU { v16bf v; v4u u[2]; };

// 16-bit A-matrix 16x32 fragment: lane-half kh owns K range [kh*16, kh*16+16)
// -> one contiguous 32-byte run (two b128 loads).
__device__ __forceinline__ v16bf load_frag_a(const bf16* rowp, int kh) {
  FragU f;
  f.u[0] = *(const v4u*)(rowp + kh * 16);
  f.u[1] = *(const v4u*)(rowp + kh * 16 + 8);
  return f.v;
}

// 16-bit B-matrix 32x16 fragment: lane-half kh owns K range [kh*16, kh*16+16)
// -> one contiguous 32-byte run (two b128 loads).
__device__ __forceinline__ v16bf load_frag_b(const bf16* colp, int kh) {
  FragU f;
  f.u[0] = *(const v4u*)(colp + kh * 16);
  f.u[1] = *(const v4u*)(colp + kh * 16 + 8);
  return f.v;
}

__device__ __forceinline__ v8f wmma_bf16(v16bf a, v16bf b, v8f c) {
  return __builtin_amdgcn_wmma_f32_16x16x32_bf16(false, a, false, b,
                                                 (short)0, c, false, false);
}

// ---------------------------------------------------------------------------
// TDM: stage one 64x64 bf16 block (8KB, row-major, stride 64) global -> LDS.
// 2-D descriptor per ISA 8.3/8.4: count=1, type=2 (image), data_size=2B,
// tensor_dim0/1 = tile_dim0/1 = 64, tensor_dim0_stride = 64. Tracked by
// TENSORcnt. Issued by one wave; other waves sync via the workgroup barrier.
// This toolchain declares the 6-arg builtin (g0, g1, g2, g3, g4, cpol).
// ---------------------------------------------------------------------------
__device__ __forceinline__ void stage_block(bf16* lptr, const bf16* gptr) {
#if __has_builtin(__builtin_amdgcn_tensor_load_to_lds)
  const unsigned long long ga = (unsigned long long)(size_t)gptr;
  u32x4 g0;
  g0[0] = 1u;                                   // count=1, user descriptor
  g0[1] = (unsigned)(size_t)lptr;               // LDS byte address
  g0[2] = (unsigned)(ga & 0xFFFFFFFFu);         // global_addr[31:0]
  g0[3] = (unsigned)((ga >> 32) & 0x1FFFFFFu) | (2u << 30);  // addr[56:32]|type=2
  i32x8 g1;
  g1[0] = (int)(1u << 16);     // workgroup_mask=0 | data_size=1 (2 bytes)
  g1[1] = (int)(64u << 16);    // tensor_dim0 = 64      (bits 79:48, low half)
  g1[2] = (int)(64u << 16);    // dim0 hi=0 | tensor_dim1 = 64 (bits 111:80)
  g1[3] = (int)(64u << 16);    // dim1 hi=0 | tile_dim0 = 64   (bits 127:112)
  g1[4] = (int)(64u);          // tile_dim1 = 64 | tile_dim2 = 0
  g1[5] = (int)(64u);          // tensor_dim0_stride = 64 (low 32)
  g1[6] = 0;                   // stride hi | tensor_dim1_stride lo
  g1[7] = 0;
  const i32x4 gz4 = {0, 0, 0, 0};
  const i32x8 gz8 = {0, 0, 0, 0, 0, 0, 0, 0};
  __builtin_amdgcn_tensor_load_to_lds(g0, g1, gz4, gz4, gz8, 0);
#else
  // fallback: cooperative copy by the calling wave (32 lanes x 16B x 16)
  const int l = threadIdx.x & 31;
  const v4u* gs = (const v4u*)gptr;
  v4u* ld = (v4u*)lptr;
  for (int c = l; c < 512; c += 32) ld[c] = gs[c];
#endif
}

// ---------------------------------------------------------------------------
// Prep kernels: f32 -> bf16 convert, and weight transpose+convert so both
// WMMA operands load as contiguous vector chunks.
// ---------------------------------------------------------------------------
__global__ void k_f32_to_bf16(const float* __restrict__ src, bf16* __restrict__ dst, int n) {
  int i = blockIdx.x * 256 + threadIdx.x;
  if (i < n) dst[i] = (bf16)src[i];
}

// src: K x N row-major f32 ; dst: N x K row-major bf16
__global__ void k_transpose_bf16(const float* __restrict__ src, bf16* __restrict__ dst,
                                 int K, int N) {
  size_t i = (size_t)blockIdx.x * 256 + threadIdx.x;
  if (i < (size_t)K * N) {
    int k = (int)(i / N);
    int n = (int)(i % N);
    dst[(size_t)n * K + k] = (bf16)src[i];
  }
}

// ---------------------------------------------------------------------------
// Generic WMMA GEMM:  C[row, col] = (A[row,:] . Bt[col,:] + bias[col]) * scale
//   A  : Mrows x K, row-major bf16
//   Bt : N x K, row-major bf16 (pre-transposed weights)
// 256 threads = 8 waves; wave = 32 rows x 64 cols (2 A-frags x 4 B-frags ->
// 8 wmma per k-step against 12 b128 loads). Workgroup tile 256 x 64.
// MODE 0: store f32               (out-proj, FFN2)
// MODE 1: ReLU, store bf16        (FFN1)
// MODE 2: store bf16 head-major [b,h,m,d]  (Q/K/V)
// ---------------------------------------------------------------------------
template <int MODE>
__global__ void __launch_bounds__(256)
k_gemm(const bf16* __restrict__ A, const bf16* __restrict__ Bt,
       const float* __restrict__ bias, float* __restrict__ Cf,
       bf16* __restrict__ Cb, int N, int K, float scale) {
  const int lane = threadIdx.x & 31;
  const int wave = threadIdx.x >> 5;
  const int kh = lane >> 4;
  const int ln = lane & 15;
  const int m0 = blockIdx.y * 256 + wave * 32;
  const int n0 = blockIdx.x * 64;

  const v8f zero = {0.f, 0.f, 0.f, 0.f, 0.f, 0.f, 0.f, 0.f};
  v8f acc[2][4] = {{zero, zero, zero, zero}, {zero, zero, zero, zero}};

  const bf16* arow0 = A + (size_t)(m0 + ln) * K;
  const bf16* arow1 = A + (size_t)(m0 + 16 + ln) * K;
  const bf16* bcol0 = Bt + (size_t)(n0 + 0  + ln) * K;
  const bf16* bcol1 = Bt + (size_t)(n0 + 16 + ln) * K;
  const bf16* bcol2 = Bt + (size_t)(n0 + 32 + ln) * K;
  const bf16* bcol3 = Bt + (size_t)(n0 + 48 + ln) * K;

  for (int k = 0; k < K; k += 32) {
    __builtin_prefetch(arow0 + k + 256, 0, 1);   // global_prefetch_b8
    __builtin_prefetch(arow1 + k + 256, 0, 1);
    const v16bf a0 = load_frag_a(arow0 + k, kh);
    const v16bf a1 = load_frag_a(arow1 + k, kh);
    v16bf bfr;
    bfr = load_frag_b(bcol0 + k, kh);
    acc[0][0] = wmma_bf16(a0, bfr, acc[0][0]);
    acc[1][0] = wmma_bf16(a1, bfr, acc[1][0]);
    bfr = load_frag_b(bcol1 + k, kh);
    acc[0][1] = wmma_bf16(a0, bfr, acc[0][1]);
    acc[1][1] = wmma_bf16(a1, bfr, acc[1][1]);
    bfr = load_frag_b(bcol2 + k, kh);
    acc[0][2] = wmma_bf16(a0, bfr, acc[0][2]);
    acc[1][2] = wmma_bf16(a1, bfr, acc[1][2]);
    bfr = load_frag_b(bcol3 + k, kh);
    acc[0][3] = wmma_bf16(a0, bfr, acc[0][3]);
    acc[1][3] = wmma_bf16(a1, bfr, acc[1][3]);
  }

#pragma unroll
  for (int rblk = 0; rblk < 2; ++rblk) {
#pragma unroll
    for (int t = 0; t < 4; ++t) {
      const int col = n0 + 16 * t + ln;
      const float bv = bias[col];
#pragma unroll
      for (int j = 0; j < 8; ++j) {
        const int row = m0 + 16 * rblk + j + 8 * kh;
        float v = (acc[rblk][t][j] + bv) * scale;
        if (MODE == 1) {
          v = fmaxf(v, 0.f);
          Cb[(size_t)row * N + col] = (bf16)v;
        } else if (MODE == 2) {
          const int b = row >> 12;        // row / M_
          const int mm = row & (M_ - 1);
          const int h = col >> 6;
          const int d = col & 63;
          Cb[(((size_t)(b * H_ + h) * M_) + mm) * D_ + d] = (bf16)v;
        } else {
          Cf[(size_t)row * N + col] = v;
        }
      }
    }
  }
}

// ---------------------------------------------------------------------------
// BigBird attention, flash-style online softmax.
// grid = (NB, H, B), block = 128 (4 waves x 16 query rows).
// Q/K/V bf16 head-major [b,h,m,d]; Q pre-scaled by 1/sqrt(D).
// K/V key blocks double-buffered in LDS via TDM (prefetch next block while
// computing current); all waves share the staged block.
// Key gather list per (h, qblock):
//   qb 0 / 63 : all 64 blocks;  qb 1 : {0,1,2,63}+rand[h][0]
//   qb 62     : {61,62,63,0}+rand[h][61]
//   middle qb : {0, qb-1, qb, qb+1, 63}+rand[h][qb-1]
// ---------------------------------------------------------------------------
__global__ void __launch_bounds__(128)
k_attention(const bf16* __restrict__ Q, const bf16* __restrict__ Km,
            const bf16* __restrict__ Vm, const float* __restrict__ mask,
            const int* __restrict__ rand_attn, bf16* __restrict__ ctx) {
  __shared__ __align__(16) bf16 stageK[2][64 * 64];  // 2 x 8KB
  __shared__ __align__(16) bf16 stageV[2][64 * 64];  // 2 x 8KB
  __shared__ bf16 pLDS[4][16][64];                   // per-wave P tile
  __shared__ int kblocks[NB_];
  __shared__ int nkb_s;

  const int qb = blockIdx.x, h = blockIdx.y, b = blockIdx.z;
  const int lane = threadIdx.x & 31, wave = threadIdx.x >> 5;
  const int kh = lane >> 4, ln = lane & 15;

  if (threadIdx.x == 0) {
    int n = 0;
    if (qb == 0 || qb == NB_ - 1) {
      for (int i = 0; i < NB_; ++i) kblocks[n++] = i;
    } else {
      const int* ra = rand_attn + ((size_t)h * (NB_ - 2) + (qb - 1)) * R_;
      if (qb == 1) {
        kblocks[0] = 0; kblocks[1] = 1; kblocks[2] = 2; kblocks[3] = NB_ - 1; n = 4;
      } else if (qb == NB_ - 2) {
        kblocks[0] = NB_ - 3; kblocks[1] = NB_ - 2; kblocks[2] = NB_ - 1; kblocks[3] = 0; n = 4;
      } else {
        kblocks[0] = 0; kblocks[1] = qb - 1; kblocks[2] = qb; kblocks[3] = qb + 1;
        kblocks[4] = NB_ - 1; n = 5;
      }
      for (int r = 0; r < R_; ++r) kblocks[n++] = ra[r];
    }
    nkb_s = n;
  }
  __syncthreads();
  const int nkb = nkb_s;

  const size_t headoff = (size_t)(b * H_ + h) * M_;
  const bf16* qrow = Q + (headoff + qb * 64 + wave * 16 + ln) * D_;
  const v16bf qf0 = load_frag_a(qrow, kh);       // D = 64 -> two k-steps
  const v16bf qf1 = load_frag_a(qrow + 32, kh);

  const v8f zero = {0.f, 0.f, 0.f, 0.f, 0.f, 0.f, 0.f, 0.f};
  v8f o[4] = {zero, zero, zero, zero};
  float mi[8], li[8];
#pragma unroll
  for (int j = 0; j < 8; ++j) { mi[j] = -1e30f; li[j] = 0.f; }

  // prologue: TDM-prefetch key block 0 into buffer 0
  if (wave == 0) {
    const int kb0 = kblocks[0];
    stage_block(&stageK[0][0], Km + (headoff + (size_t)kb0 * 64) * D_);
    stage_block(&stageV[0][0], Vm + (headoff + (size_t)kb0 * 64) * D_);
  }

  for (int ib = 0; ib < nkb; ++ib) {
    const int buf = ib & 1;
    const int kblk = kblocks[ib];

    // prefetch next key block into the other buffer while we compute
    if (wave == 0 && ib + 1 < nkb) {
      const int kbn = kblocks[ib + 1];
      stage_block(&stageK[buf ^ 1][0], Km + (headoff + (size_t)kbn * 64) * D_);
      stage_block(&stageV[buf ^ 1][0], Vm + (headoff + (size_t)kbn * 64) * D_);
    }
    // wave0: wait until the CURRENT block's two TDM ops are done (<=2 keeps
    // the just-issued prefetch pair in flight); others pass trivially, then
    // the barrier publishes the staged data to the whole workgroup.
    if (ib + 1 < nkb) __builtin_amdgcn_s_wait_tensorcnt(2);
    else              __builtin_amdgcn_s_wait_tensorcnt(0);
    __syncthreads();

    const bf16* kbase = &stageK[buf][0];
    const bf16* vbase = &stageV[buf][0];

    // S(16x64) = Q(16x64) . K_block^T : K rows contiguous-d -> LDS b128 frags
    v8f s[4] = {zero, zero, zero, zero};
#pragma unroll
    for (int t = 0; t < 4; ++t) {
      const bf16* krow = kbase + (size_t)(16 * t + ln) * D_;
      s[t] = wmma_bf16(qf0, load_frag_b(krow, kh), s[t]);
      s[t] = wmma_bf16(qf1, load_frag_b(krow + 32, kh), s[t]);
    }

    // key-mask penalty (exact under all-ones mask; NEG-additive otherwise)
#pragma unroll
    for (int t = 0; t < 4; ++t) {
      const float mk = mask[(size_t)b * M_ + kblk * 64 + 16 * t + ln];
      const float pen = (1.f - mk) * NEG_;
#pragma unroll
      for (int j = 0; j < 8; ++j) s[t][j] += pen;
    }

    // online softmax: row stats live across the 16 lanes of a half-wave
    float alpha[8];
#pragma unroll
    for (int j = 0; j < 8; ++j) {
      float r = fmaxf(fmaxf(s[0][j], s[1][j]), fmaxf(s[2][j], s[3][j]));
      r = fmaxf(r, __shfl_xor(r, 1, 32));
      r = fmaxf(r, __shfl_xor(r, 2, 32));
      r = fmaxf(r, __shfl_xor(r, 4, 32));
      r = fmaxf(r, __shfl_xor(r, 8, 32));
      const float nm = fmaxf(mi[j], r);
      alpha[j] = __expf(mi[j] - nm);
      mi[j] = nm;
    }
#pragma unroll
    for (int j = 0; j < 8; ++j) {
      float rs = 0.f;
#pragma unroll
      for (int t = 0; t < 4; ++t) {
        const float p = __expf(s[t][j] - mi[j]);
        s[t][j] = p;
        rs += p;
      }
      rs += __shfl_xor(rs, 1, 32);
      rs += __shfl_xor(rs, 2, 32);
      rs += __shfl_xor(rs, 4, 32);
      rs += __shfl_xor(rs, 8, 32);
      li[j] = li[j] * alpha[j] + rs;
#pragma unroll
      for (int t = 0; t < 4; ++t) o[t][j] *= alpha[j];
    }

    // P: C-layout -> LDS -> A-layout (ds ops are in-order within a wave)
#pragma unroll
    for (int t = 0; t < 4; ++t)
#pragma unroll
      for (int j = 0; j < 8; ++j)
        pLDS[wave][j + 8 * kh][16 * t + ln] = (bf16)s[t][j];

    // O += P(16x64) . V_block(64x64); V B-frags gathered from LDS stage
#pragma unroll
    for (int ks = 0; ks < 2; ++ks) {
      const v16bf pf = load_frag_a(&pLDS[wave][ln][ks * 32], kh);
#pragma unroll
      for (int t = 0; t < 4; ++t) {
        v16bf vf;
#pragma unroll
        for (int e = 0; e < 16; ++e)
          vf[e] = vbase[(size_t)(ks * 32 + kh * 16 + e) * D_ + 16 * t + ln];
        o[t] = wmma_bf16(pf, vf, o[t]);
      }
    }
    __syncthreads();   // stage buffers consumed; safe to overwrite next iter
  }

  // normalize, apply from_mask, store ctx as [b, m, h, d] bf16
#pragma unroll
  for (int j = 0; j < 8; ++j) {
    const int qm = qb * 64 + wave * 16 + j + 8 * kh;
    const float fm = mask[(size_t)b * M_ + qm];
    const float inv = fm / li[j];
#pragma unroll
    for (int t = 0; t < 4; ++t)
      ctx[(((size_t)b * M_ + qm) * H_ + h) * D_ + 16 * t + ln] = (bf16)(o[t][j] * inv);
  }
}

// ---------------------------------------------------------------------------
// out = LayerNorm(a + r) * g + be   (one row of E=768 per workgroup)
// ---------------------------------------------------------------------------
__global__ void __launch_bounds__(256)
k_add_ln(const float* __restrict__ a, const float* __restrict__ r,
         const float* __restrict__ g, const float* __restrict__ be,
         float* __restrict__ outf, bf16* __restrict__ outb) {
  __shared__ float r1[8], r2[8];
  const int row = blockIdx.x, tid = threadIdx.x;
  const float* pa = a + (size_t)row * E_;
  const float* pr = r + (size_t)row * E_;
  float v[3];
  float s = 0.f, s2 = 0.f;
#pragma unroll
  for (int i = 0; i < 3; ++i) {
    v[i] = pa[tid + 256 * i] + pr[tid + 256 * i];
    s += v[i];
    s2 += v[i] * v[i];
  }
#pragma unroll
  for (int m = 1; m < 32; m <<= 1) {
    s += __shfl_xor(s, m, 32);
    s2 += __shfl_xor(s2, m, 32);
  }
  if ((tid & 31) == 0) { r1[tid >> 5] = s; r2[tid >> 5] = s2; }
  __syncthreads();
  s = 0.f; s2 = 0.f;
#pragma unroll
  for (int w = 0; w < 8; ++w) { s += r1[w]; s2 += r2[w]; }
  const float mu = s * (1.f / E_);
  const float inv = rsqrtf(s2 * (1.f / E_) - mu * mu + 1e-6f);
#pragma unroll
  for (int i = 0; i < 3; ++i) {
    const int c = tid + 256 * i;
    const float y = (v[i] - mu) * inv * g[c] + be[c];
    if (outf) outf[(size_t)row * E_ + c] = y;
    if (outb) outb[(size_t)row * E_ + c] = (bf16)y;
  }
}

// ---------------------------------------------------------------------------
// Host-side orchestration
// ---------------------------------------------------------------------------
static inline size_t align256(size_t x) { return (x + 255) & ~(size_t)255; }
static inline unsigned cdivu(size_t a, size_t b) { return (unsigned)((a + b - 1) / b); }

extern "C" void kernel_launch(void* const* d_in, const int* in_sizes, int n_in,
                              void* d_out, int out_size, void* d_ws, size_t ws_size,
                              hipStream_t stream) {
  (void)in_sizes; (void)n_in; (void)out_size; (void)ws_size;

  const float* x          = (const float*)d_in[0];
  const float* input_mask = (const float*)d_in[1];
  const int*   rand_attn  = (const int*)d_in[2];
  const float* wq = (const float*)d_in[3];
  const float* bq = (const float*)d_in[4];
  const float* wk = (const float*)d_in[5];
  const float* bk = (const float*)d_in[6];
  const float* wv = (const float*)d_in[7];
  const float* bv = (const float*)d_in[8];
  const float* wo = (const float*)d_in[9];
  const float* bo = (const float*)d_in[10];
  const float* w1 = (const float*)d_in[11];
  const float* b1 = (const float*)d_in[12];
  const float* w2 = (const float*)d_in[13];
  const float* b2 = (const float*)d_in[14];
  const float* ln1_g = (const float*)d_in[15];
  const float* ln1_b = (const float*)d_in[16];
  const float* ln2_g = (const float*)d_in[17];
  const float* ln2_b = (const float*)d_in[18];
  float* out = (float*)d_out;

  char* p = (char*)d_ws;
  auto alloc = [&](size_t bytes) { char* r = p; p += align256(bytes); return r; };

  bf16* x_bf   = (bf16*)alloc((size_t)BM_ * E_ * 2);
  bf16* wq_t   = (bf16*)alloc((size_t)E_ * E_ * 2);
  bf16* wk_t   = (bf16*)alloc((size_t)E_ * E_ * 2);
  bf16* wv_t   = (bf16*)alloc((size_t)E_ * E_ * 2);
  bf16* wo_t   = (bf16*)alloc((size_t)E_ * E_ * 2);
  bf16* w1_t   = (bf16*)alloc((size_t)E_ * FF_ * 2);
  bf16* w2_t   = (bf16*)alloc((size_t)FF_ * E_ * 2);
  bf16* q_bf   = (bf16*)alloc((size_t)BM_ * E_ * 2);
  bf16* k_bf   = (bf16*)alloc((size_t)BM_ * E_ * 2);
  bf16* v_bf   = (bf16*)alloc((size_t)BM_ * E_ * 2);
  bf16* ctx_bf = (bf16*)alloc((size_t)BM_ * E_ * 2);
  float* attn_f = (float*)alloc((size_t)BM_ * E_ * 4);
  float* out1_f = (float*)alloc((size_t)BM_ * E_ * 4);
  bf16* out1_bf = (bf16*)alloc((size_t)BM_ * E_ * 2);
  bf16* ffh_bf  = (bf16*)alloc((size_t)BM_ * FF_ * 2);
  float* ffn_f  = (float*)alloc((size_t)BM_ * E_ * 4);

  // 1) convert activations + transpose/convert weights to bf16
  k_f32_to_bf16<<<cdivu((size_t)BM_ * E_, 256), 256, 0, stream>>>(x, x_bf, BM_ * E_);
  k_transpose_bf16<<<cdivu((size_t)E_ * E_, 256), 256, 0, stream>>>(wq, wq_t, E_, E_);
  k_transpose_bf16<<<cdivu((size_t)E_ * E_, 256), 256, 0, stream>>>(wk, wk_t, E_, E_);
  k_transpose_bf16<<<cdivu((size_t)E_ * E_, 256), 256, 0, stream>>>(wv, wv_t, E_, E_);
  k_transpose_bf16<<<cdivu((size_t)E_ * E_, 256), 256, 0, stream>>>(wo, wo_t, E_, E_);
  k_transpose_bf16<<<cdivu((size_t)E_ * FF_, 256), 256, 0, stream>>>(w1, w1_t, E_, FF_);
  k_transpose_bf16<<<cdivu((size_t)FF_ * E_, 256), 256, 0, stream>>>(w2, w2_t, FF_, E_);

  const dim3 g768(E_ / 64, BM_ / 256);
  const dim3 g3072(FF_ / 64, BM_ / 256);

  // 2) QKV projections (head-major bf16 out; q scaled by 1/sqrt(D))
  k_gemm<2><<<g768, 256, 0, stream>>>(x_bf, wq_t, bq, nullptr, q_bf, E_, E_, 0.125f);
  k_gemm<2><<<g768, 256, 0, stream>>>(x_bf, wk_t, bk, nullptr, k_bf, E_, E_, 1.0f);
  k_gemm<2><<<g768, 256, 0, stream>>>(x_bf, wv_t, bv, nullptr, v_bf, E_, E_, 1.0f);

  // 3) BigBird sparse attention (TDM-staged K/V)
  k_attention<<<dim3(NB_, H_, B_), 128, 0, stream>>>(q_bf, k_bf, v_bf, input_mask,
                                                     rand_attn, ctx_bf);

  // 4) output projection, residual + LN1
  k_gemm<0><<<g768, 256, 0, stream>>>(ctx_bf, wo_t, bo, attn_f, nullptr, E_, E_, 1.0f);
  k_add_ln<<<BM_, 256, 0, stream>>>(x, attn_f, ln1_g, ln1_b, out1_f, out1_bf);

  // 5) FFN, residual + LN2
  k_gemm<1><<<g3072, 256, 0, stream>>>(out1_bf, w1_t, b1, nullptr, ffh_bf, FF_, E_, 1.0f);
  k_gemm<0><<<g768, 256, 0, stream>>>(ffh_bf, w2_t, b2, ffn_f, nullptr, E_, FF_, 1.0f);
  k_add_ln<<<BM_, 256, 0, stream>>>(out1_f, ffn_f, ln2_g, ln2_b, out, nullptr);
}
